// Block_62904091017671
// MI455X (gfx1250) — matrix operand order repaired and checked by reading, not compiled, not run on hardware
//
#include <hip/hip_runtime.h>
#include <math.h>

// ---------------------------------------------------------------------------
// MI455X (gfx1250) implementation. All GEMMs use v_wmma_f32_16x16x32_bf16
// (bf16 inputs, fp32 accumulate). Problem is matrix-core bound (~150 GFLOP vs
// ~70MB of traffic => memory time ~3us @ 23.3TB/s). GEMM waves own 32x64
// output tiles so two A fragments amortize every B fragment load (8 WMMAs per
// 12 VMEM ops), keeping the matrix pipe fed; weight panels are L2-resident
// (192MB L2 >> 14MB of bf16 weights).
// ---------------------------------------------------------------------------

typedef __bf16 bf16;
typedef __attribute__((ext_vector_type(8)))  bf16  v8bf;
typedef __attribute__((ext_vector_type(16))) bf16  v16bf;
typedef __attribute__((ext_vector_type(8)))  float v8f;

// ----- problem constants ----------------------------------------------------
constexpr int Bb   = 64;
constexpr int Nn   = 198;
constexpr int Cc   = 768;
constexpr int Hh   = 12;
constexpr int HD   = 64;       // head dim
constexpr int NKV  = 3 * Cc;   // 2304
constexpr int CH   = 4 * Cc;   // 3072
constexpr int NPAD  = 208;     // 13*16  (padded seq for S tiles)
constexpr int NPAD2 = 224;     // 7*32   (padded seq for P@V K loop)
constexpr int NTILE = 13;
constexpr int KKEEP = 138;     // ceil(0.7*196)
constexpr int NOUT  = 141;     // 2 + K + 1
constexpr int M1 = Bb * Nn;    // 12672 rows (pre-prune)
constexpr int M2 = Bb * NOUT;  // 9024 rows (post-prune)

// ----- helpers --------------------------------------------------------------
__device__ inline v8f zero8() {
  v8f z = {0.f,0.f,0.f,0.f,0.f,0.f,0.f,0.f};
  return z;
}

__device__ inline bf16 f2bf(float f) {          // RNE float -> bf16
  unsigned u = __float_as_uint(f);
  unsigned r = u + 0x7FFFu + ((u >> 16) & 1u);
  unsigned short h = (unsigned short)(r >> 16);
  return __builtin_bit_cast(bf16, h);
}

// A fragment: 16x32 bf16 tile, row-major source (lda in elements).
// Contiguous-k order used here: lane holds row m=lane&15; element e -> k = half*16 + e.
// Two contiguous 16B loads per lane.
__device__ inline v16bf load_a16(const bf16* A, int lda, int k0) {
  int lane = threadIdx.x & 31;
  int m  = lane & 15;
  int hf = lane >> 4;
  const bf16* p = A + (size_t)m * lda + k0 + hf * 16;
  v8bf lo = *(const v8bf*)(p);
  v8bf hi = *(const v8bf*)(p + 8);
  return __builtin_shufflevector(lo, hi, 0,1,2,3,4,5,6,7,8,9,10,11,12,13,14,15);
}

// B fragment from B^T (N x K row-major, ldk in elements), tile col base n0
// already folded into BT pointer. Contiguous-k order used here: lane holds col n=lane&15,
// element e -> k = half*16 + e.  One contiguous 32B load per lane.
__device__ inline v16bf load_bt16(const bf16* BT, int ldk, int k0) {
  int lane = threadIdx.x & 31;
  int n  = lane & 15;
  int hf = lane >> 4;
  return *(const v16bf*)(BT + (size_t)n * ldk + k0 + hf * 16);
}

__device__ inline v8f wmma_bf16(v16bf a, v16bf b, v8f c) {
  return __builtin_amdgcn_wmma_f32_16x16x32_bf16(false, a, false, b,
                                                 (short)0, c, false, false);
}

// ---------------------------------------------------------------------------
// Weight convert: fp32 (K x N) row-major  ->  bf16 (N x K) row-major (B^T)
// ---------------------------------------------------------------------------
__global__ __launch_bounds__(256)
void transpose_to_bf16(const float* __restrict__ in, bf16* __restrict__ out,
                       int K, int N) {
  __shared__ float tile[32][33];
  int n0 = blockIdx.x * 32, k0 = blockIdx.y * 32;
  int tx = threadIdx.x & 31, ty = threadIdx.x >> 5;   // 32 x 8
  for (int i = ty; i < 32; i += 8) {
    int k = k0 + i, n = n0 + tx;
    tile[i][tx] = (k < K && n < N) ? in[(size_t)k * N + n] : 0.f;
  }
  __syncthreads();
  for (int i = ty; i < 32; i += 8) {
    int n = n0 + i, k = k0 + tx;
    if (n < N && k < K) out[(size_t)n * K + k] = f2bf(tile[tx][i]);
  }
}

// ---------------------------------------------------------------------------
// LayerNorm over C=768, one 256-thread block per row, bf16 output
// ---------------------------------------------------------------------------
__global__ __launch_bounds__(256)
void ln_bf16(const float* __restrict__ x, const float* __restrict__ g,
             const float* __restrict__ bt, bf16* __restrict__ out) {
  __shared__ float red[256];
  int row = blockIdx.x, tid = threadIdx.x;
  const float* xr = x + (size_t)row * Cc;
  float v0 = xr[tid], v1 = xr[tid + 256], v2 = xr[tid + 512];
  red[tid] = v0 + v1 + v2;
  __syncthreads();
  for (int off = 128; off > 0; off >>= 1) {
    if (tid < off) red[tid] += red[tid + off];
    __syncthreads();
  }
  float mu = red[0] * (1.f / 768.f);
  __syncthreads();
  float d0 = v0 - mu, d1 = v1 - mu, d2 = v2 - mu;
  red[tid] = d0 * d0 + d1 * d1 + d2 * d2;
  __syncthreads();
  for (int off = 128; off > 0; off >>= 1) {
    if (tid < off) red[tid] += red[tid + off];
    __syncthreads();
  }
  float rstd = rsqrtf(red[0] * (1.f / 768.f) + 1e-5f);
  bf16* orow = out + (size_t)row * Cc;
  orow[tid]       = f2bf(d0 * rstd * g[tid]       + bt[tid]);
  orow[tid + 256] = f2bf(d1 * rstd * g[tid + 256] + bt[tid + 256]);
  orow[tid + 512] = f2bf(d2 * rstd * g[tid + 512] + bt[tid + 512]);
}

// ---------------------------------------------------------------------------
// Generic bf16 WMMA GEMM:  out(M x N) = A(M x K) @ B(K x N) + bias, epilogues:
//   MODE 0: fp32 out            MODE 1: fp32 out + residual
//   MODE 2: bf16 out, exact GELU MODE 3: bf16 scatter into q/k/vT head layout
// 256 threads = 8 waves; each wave owns a 32x64 tile (2 A frags x 4 B frags,
// 8 accumulators = 64 VGPRs). Block covers 256 x 64.
// Requires M%32==0, N%64==0, K%32==0 (true for all call sites).
// ---------------------------------------------------------------------------
template <int MODE>
__global__ __launch_bounds__(256)
void gemm_bf16(int M, int N, int K,
               const bf16* __restrict__ A, int lda,
               const bf16* __restrict__ BT, int ldb,
               const float* __restrict__ bias,
               float* __restrict__ outf, const float* __restrict__ res,
               bf16* __restrict__ outb,
               bf16* __restrict__ oq, bf16* __restrict__ ok,
               bf16* __restrict__ ovt) {
  int wave = threadIdx.x >> 5;
  int lane = threadIdx.x & 31;
  int n16 = lane & 15, hf = lane >> 4;
  int row0 = blockIdx.y * 256 + wave * 32;
  int col0 = blockIdx.x * 64;
  if (row0 >= M) return;

  v8f acc[2][4];
#pragma unroll
  for (int mt = 0; mt < 2; mt++)
#pragma unroll
    for (int nt = 0; nt < 4; nt++) acc[mt][nt] = zero8();

  const bf16* A0 = A + (size_t)row0 * lda;
  const bf16* A1 = A + (size_t)(row0 + 16) * lda;
  for (int k0 = 0; k0 < K; k0 += 32) {
    v16bf a0 = load_a16(A0, lda, k0);
    v16bf a1 = load_a16(A1, lda, k0);
#pragma unroll
    for (int nt = 0; nt < 4; nt++) {
      v16bf bf = load_bt16(BT + (size_t)(col0 + nt * 16) * ldb, ldb, k0);
      acc[0][nt] = wmma_bf16(a0, bf, acc[0][nt]);
      acc[1][nt] = wmma_bf16(a1, bf, acc[1][nt]);
    }
  }

#pragma unroll
  for (int mt = 0; mt < 2; mt++) {
#pragma unroll
    for (int nt = 0; nt < 4; nt++) {
#pragma unroll
      for (int r = 0; r < 8; r++) {
        int row = row0 + mt * 16 + hf * 8 + r;
        int col = col0 + nt * 16 + n16;
        float v = acc[mt][nt][r] + bias[col];
        size_t idx = (size_t)row * N + col;
        if (MODE == 0) {
          outf[idx] = v;
        } else if (MODE == 1) {
          outf[idx] = v + res[idx];
        } else if (MODE == 2) {
          outb[idx] = f2bf(0.5f * v * (1.f + erff(v * 0.70710678f)));  // exact GELU
        } else {                     // QKV scatter: col = w*768 + h*64 + d
          int w = col / Cc, rem = col % Cc, hh = rem / HD, d = rem % HD;
          int b = row / Nn, n = row % Nn;
          size_t bh = (size_t)b * Hh + hh;
          bf16 bv = f2bf(v);
          if (w == 0)      oq [(bh * NPAD + n) * HD + d]   = bv;
          else if (w == 1) ok [(bh * NPAD + n) * HD + d]   = bv;
          else             ovt[(bh * HD + d) * NPAD2 + n]  = bv;
        }
      }
    }
  }
}

// ---------------------------------------------------------------------------
// Attention: one wave per (b, h, 16-row tile). S = q kT * 1/8, softmax,
// out = P v.  P re-fragmented via LDS.  Rows 0/1 (CLS & "priv") written
// per-head for deterministic reduction in the prune kernel.
// ---------------------------------------------------------------------------
__global__ __launch_bounds__(32)
void attn_kernel(const bf16* __restrict__ q,   // (B*H, NPAD, 64)
                 const bf16* __restrict__ k,   // (B*H, NPAD, 64)
                 const bf16* __restrict__ vt,  // (B*H, 64, NPAD2)
                 bf16* __restrict__ out,       // (B, 198, 768)
                 float* __restrict__ rowbuf) { // (B*H, 2, 196)
  int bid = blockIdx.x;
  int rt = bid % NTILE;
  int bh = bid / NTILE;
  int b = bh / Hh, h = bh % Hh;
  int lane = threadIdx.x;
  int n16 = lane & 15, hf = lane >> 4;

  __shared__ bf16 P[16 * NPAD2];

  const bf16* qh = q + (size_t)bh * NPAD * HD + (size_t)rt * 16 * HD;
  const bf16* kh = k + (size_t)bh * NPAD * HD;

  v8f acc[NTILE];
#pragma unroll
  for (int t = 0; t < NTILE; t++) acc[t] = zero8();

  v16bf a0 = load_a16(qh, HD, 0);
  v16bf a1 = load_a16(qh, HD, 32);
#pragma unroll
  for (int t = 0; t < NTILE; t++) {
    v16bf b0 = load_bt16(kh + (size_t)t * 16 * HD, HD, 0);
    v16bf b1 = load_bt16(kh + (size_t)t * 16 * HD, HD, 32);
    acc[t] = wmma_bf16(a0, b0, acc[t]);
    acc[t] = wmma_bf16(a1, b1, acc[t]);
  }

  const float scale = 0.125f;  // HD^-0.5
#pragma unroll
  for (int t = 0; t < NTILE; t++)
#pragma unroll
    for (int r = 0; r < 8; r++) acc[t][r] *= scale;
  if (n16 >= 6) {              // cols 198..207 invalid (tile 12)
#pragma unroll
    for (int r = 0; r < 8; r++) acc[NTILE - 1][r] = -1e30f;
  }

  // softmax per row (row m = hf*8 + r lives on the 16 lanes of this half)
#pragma unroll
  for (int r = 0; r < 8; r++) {
    float m = -1e30f;
#pragma unroll
    for (int t = 0; t < NTILE; t++) m = fmaxf(m, acc[t][r]);
    for (int off = 1; off < 16; off <<= 1) m = fmaxf(m, __shfl_xor(m, off));
    float s = 0.f;
#pragma unroll
    for (int t = 0; t < NTILE; t++) {
      float e = __expf(acc[t][r] - m);
      acc[t][r] = e;
      s += e;
    }
    for (int off = 1; off < 16; off <<= 1) s += __shfl_xor(s, off);
    float inv = 1.f / s;
#pragma unroll
    for (int t = 0; t < NTILE; t++) acc[t][r] *= inv;
  }

  // CLS (row 0) and privacy (row 1) attention rows, per head
  if (rt == 0 && hf == 0) {
#pragma unroll
    for (int r = 0; r < 2; r++) {
#pragma unroll
      for (int t = 0; t < NTILE; t++) {
        int col = t * 16 + n16;
        if (col >= 2 && col < Nn)
          rowbuf[((size_t)bh * 2 + r) * 196 + (col - 2)] = acc[t][r];
      }
    }
  }

  // stage P (bf16) to LDS, zero-pad cols 208..223
#pragma unroll
  for (int r = 0; r < 8; r++) {
    int row = hf * 8 + r;
#pragma unroll
    for (int t = 0; t < NTILE; t++)
      P[row * NPAD2 + t * 16 + n16] = f2bf(acc[t][r]);
    P[row * NPAD2 + NPAD + n16] = f2bf(0.f);
  }
  __syncthreads();

  // out strip = P (16 x 224) @ v (224 x 64)
  v8f oacc[4];
#pragma unroll
  for (int nt = 0; nt < 4; nt++) oacc[nt] = zero8();
  const bf16* vth = vt + (size_t)bh * HD * NPAD2;
#pragma unroll
  for (int kt = 0; kt < NPAD2 / 32; kt++) {
    v16bf af = load_a16((const bf16*)P, NPAD2, kt * 32);
#pragma unroll
    for (int nt = 0; nt < 4; nt++) {
      v16bf bf = load_bt16(vth + (size_t)nt * 16 * NPAD2, NPAD2, kt * 32);
      oacc[nt] = wmma_bf16(af, bf, oacc[nt]);
    }
  }

#pragma unroll
  for (int nt = 0; nt < 4; nt++)
#pragma unroll
    for (int r = 0; r < 8; r++) {
      int row = rt * 16 + hf * 8 + r;
      if (row < Nn)
        out[((size_t)b * Nn + row) * Cc + h * HD + nt * 16 + n16] =
            f2bf(oacc[nt][r]);
    }
}

// ---------------------------------------------------------------------------
// Token prune: per batch — normalize act/priv, score, rank (== lax.top_k
// order: value desc, index asc on ties), gather kept tokens, build fused
// extra token, emit x2 (B, 141, 768).
// ---------------------------------------------------------------------------
__global__ __launch_bounds__(256)
void prune_kernel(const float* __restrict__ x1,
                  const float* __restrict__ rowbuf,
                  float* __restrict__ x2) {
  int b = blockIdx.x, tid = threadIdx.x;
  __shared__ float sa[196], sc[196], red[256];
  __shared__ int rk[196];

  float a = 0.f, p = 0.f;
  if (tid < 196) {
    for (int h = 0; h < Hh; h++) {
      size_t base = ((size_t)(b * Hh + h)) * 2 * 196;
      a += rowbuf[base + tid];
      p += rowbuf[base + 196 + tid];
    }
  }
  red[tid] = (tid < 196) ? a : 0.f;
  __syncthreads();
  for (int off = 128; off > 0; off >>= 1) {
    if (tid < off) red[tid] += red[tid + off];
    __syncthreads();
  }
  float asum = red[0];
  __syncthreads();
  red[tid] = (tid < 196) ? p : 0.f;
  __syncthreads();
  for (int off = 128; off > 0; off >>= 1) {
    if (tid < off) red[tid] += red[tid + off];
    __syncthreads();
  }
  float psum = red[0];
  __syncthreads();

  if (tid < 196) {
    float an = a / asum;
    sa[tid] = an;
    sc[tid] = an - 0.5f * (p / psum);
  }
  __syncthreads();
  if (tid < 196) {
    float si = sc[tid];
    int r = 0;
    for (int j = 0; j < 196; j++) {
      float sj = sc[j];
      r += (sj > si) || (sj == si && j < tid);
    }
    rk[tid] = r;
  }
  __syncthreads();

  const float* xb = x1 + (size_t)b * Nn * Cc;
  float* yb = x2 + (size_t)b * NOUT * Cc;
  for (int c = tid; c < 2 * Cc; c += 256) yb[c] = xb[c];  // CLS + token 1

  float e0 = 0.f, e1 = 0.f, e2 = 0.f;
  for (int i = 0; i < 196; i++) {
    const float* src = xb + (size_t)(2 + i) * Cc;
    int r = rk[i];
    if (r < KKEEP) {
      float* dst = yb + (size_t)(2 + r) * Cc;
      for (int c = tid; c < Cc; c += 256) dst[c] = src[c];
    } else {
      float w = sa[i];
      e0 += w * src[tid];
      e1 += w * src[tid + 256];
      e2 += w * src[tid + 512];
    }
  }
  float* er = yb + (size_t)(NOUT - 1) * Cc;
  er[tid] = e0;
  er[tid + 256] = e1;
  er[tid + 512] = e2;
}

// ---------------------------------------------------------------------------
extern "C" void kernel_launch(void* const* d_in, const int* in_sizes, int n_in,
                              void* d_out, int out_size, void* d_ws,
                              size_t ws_size, hipStream_t stream) {
  const float* x      = (const float*)d_in[0];
  const float* ln1_g  = (const float*)d_in[1];
  const float* ln1_b  = (const float*)d_in[2];
  const float* qkv_w  = (const float*)d_in[3];
  const float* qkv_b  = (const float*)d_in[4];
  const float* proj_w = (const float*)d_in[5];
  const float* proj_b = (const float*)d_in[6];
  const float* ln2_g  = (const float*)d_in[7];
  const float* ln2_b  = (const float*)d_in[8];
  const float* fc1_w  = (const float*)d_in[9];
  const float* fc1_b  = (const float*)d_in[10];
  const float* fc2_w  = (const float*)d_in[11];
  const float* fc2_b  = (const float*)d_in[12];

  char* ws = (char*)d_ws;
  size_t off = 0;
  auto carve = [&](size_t bytes) -> char* {
    char* p = ws + off;
    off += (bytes + 255) & ~(size_t)255;
    return p;
  };

  bf16*  qkvWT  = (bf16*)carve((size_t)NKV * Cc * 2);
  bf16*  projWT = (bf16*)carve((size_t)Cc * Cc * 2);
  bf16*  fc1WT  = (bf16*)carve((size_t)CH * Cc * 2);
  bf16*  fc2WT  = (bf16*)carve((size_t)Cc * CH * 2);
  bf16*  h1     = (bf16*)carve((size_t)M1 * Cc * 2);
  size_t qbytes = (size_t)Bb * Hh * NPAD * HD * 2;
  size_t vbytes = (size_t)Bb * Hh * HD * NPAD2 * 2;
  bf16*  qb     = (bf16*)carve(qbytes);
  bf16*  kb     = (bf16*)carve(qbytes);
  bf16*  vtb    = (bf16*)carve(vbytes);
  bf16*  attnout= (bf16*)carve((size_t)M1 * Cc * 2);
  float* x1     = (float*)carve((size_t)M1 * Cc * 4);
  float* rowbuf = (float*)carve((size_t)Bb * Hh * 2 * 196 * 4);
  float* x2     = (float*)carve((size_t)M2 * Cc * 4);
  bf16*  h2     = (bf16*)carve((size_t)M2 * Cc * 2);
  bf16*  hmid   = (bf16*)carve((size_t)M2 * CH * 2);
  (void)ws_size; (void)in_sizes; (void)n_in; (void)out_size;

  dim3 tb(256);
  // weights -> bf16 B^T
  transpose_to_bf16<<<dim3(NKV / 32, Cc / 32), tb, 0, stream>>>(qkv_w,  qkvWT,  Cc, NKV);
  transpose_to_bf16<<<dim3(Cc  / 32, Cc / 32), tb, 0, stream>>>(proj_w, projWT, Cc, Cc);
  transpose_to_bf16<<<dim3(CH  / 32, Cc / 32), tb, 0, stream>>>(fc1_w,  fc1WT,  Cc, CH);
  transpose_to_bf16<<<dim3(Cc  / 32, CH / 32), tb, 0, stream>>>(fc2_w,  fc2WT,  CH, Cc);

  // LN1 -> bf16
  ln_bf16<<<M1, tb, 0, stream>>>(x, ln1_g, ln1_b, h1);

  // zero q/k/vT pads, then QKV GEMM scattering into head layouts
  hipMemsetAsync(qb,  0, qbytes, stream);
  hipMemsetAsync(kb,  0, qbytes, stream);
  hipMemsetAsync(vtb, 0, vbytes, stream);
  gemm_bf16<3><<<dim3(NKV / 64, (M1 + 255) / 256), tb, 0, stream>>>(
      M1, NKV, Cc, h1, Cc, qkvWT, Cc, qkv_b,
      nullptr, nullptr, nullptr, qb, kb, vtb);

  // attention
  attn_kernel<<<Bb * Hh * NTILE, 32, 0, stream>>>(qb, kb, vtb, attnout, rowbuf);

  // proj + residual -> x1 (fp32)
  gemm_bf16<1><<<dim3(Cc / 64, (M1 + 255) / 256), tb, 0, stream>>>(
      M1, Cc, Cc, attnout, Cc, projWT, Cc, proj_b,
      x1, x, nullptr, nullptr, nullptr, nullptr);

  // token selection -> x2 (B, 141, 768)
  prune_kernel<<<Bb, tb, 0, stream>>>(x1, rowbuf, x2);

  // LN2 -> bf16
  ln_bf16<<<M2, tb, 0, stream>>>(x2, ln2_g, ln2_b, h2);

  // MLP
  gemm_bf16<2><<<dim3(CH / 64, (M2 + 255) / 256), tb, 0, stream>>>(
      M2, CH, Cc, h2, Cc, fc1WT, Cc, fc1_b,
      nullptr, nullptr, hmid, nullptr, nullptr, nullptr);
  gemm_bf16<1><<<dim3(Cc / 64, (M2 + 255) / 256), tb, 0, stream>>>(
      M2, Cc, CH, hmid, CH, fc2WT, CH, fc2_b,
      (float*)d_out, x2, nullptr, nullptr, nullptr, nullptr);
}
